// MyriadStepByStep_55173149884703
// MI455X (gfx1250) — matrix operand order repaired and checked by reading, not compiled, not run on hardware
//
#include <hip/hip_runtime.h>
#include <hip/hip_bf16.h>
#include <math.h>

// ---------------------------------------------------------------------------
// Problem constants (from the reference)
// ---------------------------------------------------------------------------
#define B_    2
#define H_    16
#define S_    2048
#define D_    72
#define DP_   96        // D padded to 3 x 32 for WMMA K-chunks
#define LPRE  576
#define LSEQ  512
#define NDENSE ((LPRE + LSEQ) / 32)   // 34 dense key tiles for tail queries

typedef __attribute__((ext_vector_type(16))) _Float16 v16h;
typedef __attribute__((ext_vector_type(8)))  _Float16 v8h;
typedef __attribute__((ext_vector_type(8)))  float    v8f;

template <bool V> struct BoolC { static constexpr bool value = V; };

struct KFrag { v16h f[6]; };   // 2 key halves x 3 D-chunks
struct VFrag { v16h f[5]; };   // 5 D-tiles (d padded to 80; col 72 = ones)

static __device__ __forceinline__ v16h cat8(v8h lo, v8h hi) {
  v16h r;
#pragma unroll
  for (int i = 0; i < 8; ++i) { r[i] = lo[i]; r[i + 8] = hi[i]; }
  return r;
}

static __device__ __forceinline__ v8f wmma_f16(v16h a, v16h b, v8f c) {
  // D = A(16x32 f16) * B(32x16 f16) + C(16x16 f32)
  return __builtin_amdgcn_wmma_f32_16x16x32_f16(
      /*neg_a=*/false, a, /*neg_b=*/false, b,
      /*c_mod=*/(short)0, c, /*reuse_a=*/false, /*reuse_b=*/false);
}

// ---------------------------------------------------------------------------
// Kernel 1: rotary + L2-normalize + pack to f16 (WMMA-friendly layouts)
//   qh[bh][s][96], kh[bh][s][96]  (row-major, zero-padded d=72..95)
//   vt[bh][96][S]                 (V transposed; row 72 = 1.0 so the 5th PV
//                                  WMMA also accumulates the softmax denom)
// One wave per (b,h,s) row; lanes cover d = {lane, lane+32, lane+64}.
// ---------------------------------------------------------------------------
__global__ void __launch_bounds__(128)
rope_norm_pack(const float* __restrict__ q, const float* __restrict__ k,
               const float* __restrict__ v, const float* __restrict__ pos,
               const float* __restrict__ pos_o, const float* __restrict__ tim,
               const float* __restrict__ freqs, const float* __restrict__ tf,
               const float* __restrict__ scale,
               _Float16* __restrict__ qh, _Float16* __restrict__ kh,
               _Float16* __restrict__ vt)
{
  const int wave = blockIdx.x * (blockDim.x >> 5) + (threadIdx.x >> 5);
  const int lane = threadIdx.x & 31;
  const int s  = wave % S_;
  const int bh = wave / S_;
  const int h  = bh % H_;
  const int b  = bh / H_;
  const size_t row = (size_t)bh * S_ + s;
  const float* qr = q + row * D_;
  const float* kr = k + row * D_;
  const float* vr = v + row * D_;

  const float px_o = pos_o[((size_t)b * S_ + s) * 2 + 0] * 2.f - 1.f;
  const float py_o = pos_o[((size_t)b * S_ + s) * 2 + 1] * 2.f - 1.f;
  const float px_n = pos  [((size_t)b * S_ + s) * 2 + 0] * 2.f - 1.f;
  const float py_n = pos  [((size_t)b * S_ + s) * 2 + 1] * 2.f - 1.f;
  const float tm   = tim[(size_t)b * S_ + s];

  // theta(j), j in [0,30): [th_h(orig), th_h(new), th_w(orig), th_w(new), th_t]
  auto theta = [&](int j) -> float {
    if (j < 6)  return py_o * freqs[96 + h * 6 + j];        // freqs[1][h][j]
    if (j < 12) return py_n * freqs[96 + h * 6 + (j - 6)];
    if (j < 18) return px_o * freqs[h * 6 + (j - 12)];      // freqs[0][h][.]
    if (j < 24) return px_n * freqs[h * 6 + (j - 18)];
    return tm * tf[h * 6 + (j - 24)];
  };

  float rq[3], rk[3];
  const int idxs[3] = {lane, lane + 32, lane + 64};
#pragma unroll
  for (int e = 0; e < 3; ++e) {
    const int i = idxs[e];
    float vq = 0.f, vk = 0.f;
    if (i < D_) {
      if (i >= 60) { vq = qr[i]; vk = kr[i]; }
      else {
        const int j = (i < 30) ? i : i - 30;
        const float th = theta(j);
        const float c = __cosf(th), sn = __sinf(th);
        if (i < 30) { vq = qr[i] * c - qr[i + 30] * sn;
                      vk = kr[i] * c - kr[i + 30] * sn; }
        else        { vq = qr[i] * c + qr[i - 30] * sn;
                      vk = kr[i] * c + kr[i - 30] * sn; }
      }
    }
    rq[e] = vq; rk[e] = vk;
  }

  float sq = rq[0] * rq[0] + rq[1] * rq[1] + rq[2] * rq[2];
  float sk = rk[0] * rk[0] + rk[1] * rk[1] + rk[2] * rk[2];
#pragma unroll
  for (int m = 1; m < 32; m <<= 1) {
    sq += __shfl_xor(sq, m, 32);
    sk += __shfl_xor(sk, m, 32);
  }
  const float ss = __fsqrt_rn(scale[h]);
  const float nq = ss * __frsqrt_rn(sq + 1e-6f);
  const float nk = ss * __frsqrt_rn(sk + 1e-6f);

  _Float16* qhr = qh + row * DP_;
  _Float16* khr = kh + row * DP_;
  _Float16* vtb = vt + (size_t)bh * DP_ * S_;
#pragma unroll
  for (int e = 0; e < 3; ++e) {
    const int i = idxs[e];
    qhr[i] = (_Float16)(rq[e] * nq);
    khr[i] = (_Float16)(rk[e] * nk);
    // d<72: value; d==72: ones column (softmax-denominator trick); else 0
    const float vv = (i < D_) ? vr[i] : ((i == D_) ? 1.f : 0.f);
    vtb[(size_t)i * S_ + s] = (_Float16)vv;
  }
}

// ---------------------------------------------------------------------------
// Kernel 2: flash attention, one wave per 16-query tile.
// Software-pipelined: K fragments for tile t+1 are loaded right after the
// score WMMAs of tile t (softmax + 5 PV WMMAs hide the latency); V fragments
// load at iteration top and are consumed last. Ones-column in V^T makes
// O[:,72] the softmax denominator (no row-sum shuffles).
// ---------------------------------------------------------------------------
__global__ void __launch_bounds__(128)
flash_attn(const _Float16* __restrict__ qh, const _Float16* __restrict__ kh,
           const _Float16* __restrict__ vt, float* __restrict__ out)
{
  __shared__ _Float16 pbuf[4][16][32];   // per-wave P staging (C->A relayout)

  const int w    = threadIdx.x >> 5;
  const int lane = threadIdx.x & 31;
  const int job  = blockIdx.x * 4 + w;       // B*H*(S/16) = 4096 jobs
  const int qt   = job & 127;
  const int bh   = job >> 7;
  const int qbase = qt * 16;

  float* orow = out + ((size_t)bh * S_ + qbase) * D_;

  if (qbase + 16 <= LPRE) {                  // fully masked prefix queries
    for (int t = lane; t < 16 * D_; t += 32) orow[t] = 0.f;
    return;
  }

  const _Float16* qrow  = qh + ((size_t)bh * S_ + qbase) * DP_;
  const _Float16* krow  = kh + (size_t)bh * S_ * DP_;
  const _Float16* vbase = vt + (size_t)bh * DP_ * S_;

  const int rlo  = lane & 15;                // A-frag row / B-frag column
  const int hk   = (lane & 16) ? 16 : 0;     // A-frag K sub-offset
  const int koff = (lane & 16) ? 16 : 0;     // B-frag K sub-offset
  const int row0 = (lane & 16) >> 1;         // C-frag row base (0 or 8)

  // Q A-fragments for the three 32-wide D chunks (padding d>=72 is zero)
  v16h aq0, aq1, aq2;
  {
    const _Float16* p = qrow + rlo * DP_ + hk;
    aq0 = cat8(*(const v8h*)(p +  0), *(const v8h*)(p +  8));
    aq1 = cat8(*(const v8h*)(p + 32), *(const v8h*)(p + 40));
    aq2 = cat8(*(const v8h*)(p + 64), *(const v8h*)(p + 72));
  }

  float m_[8];
  v8f o0 = {}, o1 = {}, o2 = {}, o3 = {}, o4 = {};
#pragma unroll
  for (int r = 0; r < 8; ++r) m_[r] = -1e30f;

  // ---- fragment loaders (loop-carried pointers, immediate offsets) ----
  auto load_k = [&](const _Float16* p, KFrag& kf) {
    kf.f[0] = *(const v16h*)(p);
    kf.f[1] = *(const v16h*)(p + 16 * DP_);
    kf.f[2] = *(const v16h*)(p + 32);
    kf.f[3] = *(const v16h*)(p + 16 * DP_ + 32);
    kf.f[4] = *(const v16h*)(p + 64);
    kf.f[5] = *(const v16h*)(p + 16 * DP_ + 64);
  };
  auto load_v = [&](const _Float16* p, VFrag& vf) {
#pragma unroll
    for (int j = 0; j < 5; ++j) vf.f[j] = *(const v16h*)(p + (size_t)j * 16 * S_);
  };
  auto scores = [&](const KFrag& kf, v8f& c_lo, v8f& c_hi) {
    c_lo = wmma_f16(aq0, kf.f[0], c_lo);
    c_hi = wmma_f16(aq0, kf.f[1], c_hi);
    c_lo = wmma_f16(aq1, kf.f[2], c_lo);
    c_hi = wmma_f16(aq1, kf.f[3], c_hi);
    c_lo = wmma_f16(aq2, kf.f[4], c_lo);
    c_hi = wmma_f16(aq2, kf.f[5], c_hi);
  };

  auto softmax_pv = [&](int kbase, v8f c_lo, v8f c_hi, const VFrag& vf,
                        auto masked_tag) {
    constexpr bool MASKED = decltype(masked_tag)::value;
    const int qn0   = qbase + row0 - LPRE;
    const int kn_lo = kbase + rlo - LPRE;
    const int kn_hi = kn_lo + 16;
    float alpha[8];
#pragma unroll
    for (int r = 0; r < 8; ++r) {
      float slo = c_lo[r];
      float shi = c_hi[r];
      if (MASKED) {
        const int qn = qn0 + r;
        const bool alo = (qn == kn_lo) | ((qn >= kn_lo) & (qn < LSEQ)) |
                         ((kn_lo < LSEQ) & (qn >= LSEQ));
        const bool ahi = (qn == kn_hi) | ((qn >= kn_hi) & (qn < LSEQ)) |
                         ((kn_hi < LSEQ) & (qn >= LSEQ));
        slo = alo ? slo : -1e30f;
        shi = ahi ? shi : -1e30f;
      }
      float mx = fmaxf(slo, shi);
      mx = fmaxf(mx, __shfl_xor(mx, 1));
      mx = fmaxf(mx, __shfl_xor(mx, 2));
      mx = fmaxf(mx, __shfl_xor(mx, 4));
      mx = fmaxf(mx, __shfl_xor(mx, 8));
      const float mn = fmaxf(m_[r], mx);
      alpha[r] = __expf(m_[r] - mn);
      m_[r] = mn;
      const float plo = __expf(slo - mn);
      const float phi = __expf(shi - mn);
      pbuf[w][row0 + r][rlo]      = (_Float16)plo;   // C-layout -> LDS
      pbuf[w][row0 + r][rlo + 16] = (_Float16)phi;
    }
#pragma unroll
    for (int r = 0; r < 8; ++r) {
      o0[r] *= alpha[r]; o1[r] *= alpha[r]; o2[r] *= alpha[r];
      o3[r] *= alpha[r]; o4[r] *= alpha[r];
    }
    // reload P as a 16x32 A-fragment (DS ops in-order per wave)
    const v16h ap = cat8(*(const v8h*)&pbuf[w][rlo][hk],
                         *(const v8h*)&pbuf[w][rlo][8 + hk]);
    o0 = wmma_f16(ap, vf.f[0], o0);
    o1 = wmma_f16(ap, vf.f[1], o1);
    o2 = wmma_f16(ap, vf.f[2], o2);
    o3 = wmma_f16(ap, vf.f[3], o3);
    o4 = wmma_f16(ap, vf.f[4], o4);
  };

  // ---- key-tile schedule from the analytic mask structure ----
  // causal block (576 <= q < 1088): tiles with kbase+31 <= qbase are unmasked;
  // tail block (q >= 1088): all 34 dense tiles unmasked + diagonal-only tile.
  int n_full, n_kt;
  if (qbase < LPRE + LSEQ) {
    n_full = ((qbase - 31) >> 5) + 1;          // >= 18, so always >= 1
    n_kt   = (qbase + 16 + 31) >> 5;
  } else {
    n_full = NDENSE;
    n_kt   = NDENSE;
  }

  const _Float16* kpt = krow + (size_t)rlo * DP_ + koff;
  const _Float16* vpt = vbase + (size_t)rlo * S_ + koff;

  KFrag kf;
  VFrag vf;
  load_k(kpt, kf);                              // prime the pipeline
  for (int kt = 0; kt < n_full; ++kt) {
    load_v(vpt, vf);                            // consumed at iteration end
    v8f c_lo = {}, c_hi = {};
    scores(kf, c_lo, c_hi);                     // consumes previous K loads
    kpt += 32 * DP_;
    if (kt + 1 < n_full) load_k(kpt, kf);       // preload next tile's K
    softmax_pv(kt * 32, c_lo, c_hi, vf, BoolC<false>{});
    vpt += 32;
  }
  for (int kt = n_full; kt < n_kt; ++kt) {      // masked boundary tiles
    load_k(kpt, kf);
    load_v(vpt, vf);
    v8f c_lo = {}, c_hi = {};
    scores(kf, c_lo, c_hi);
    softmax_pv(kt * 32, c_lo, c_hi, vf, BoolC<true>{});
    kpt += 32 * DP_;
    vpt += 32;
  }
  if (qbase >= LPRE + LSEQ) {                   // diagonal-only tile
    const int kb2 = (qbase >> 5) << 5;
    load_k(krow + (size_t)(kb2 + rlo) * DP_ + koff, kf);
    load_v(vbase + (size_t)rlo * S_ + kb2 + koff, vf);
    v8f c_lo = {}, c_hi = {};
    scores(kf, c_lo, c_hi);
    softmax_pv(kb2, c_lo, c_hi, vf, BoolC<true>{});
  }

  // ---- epilogue: divide by denominator (O column d==72), D=72 outputs ----
#pragma unroll
  for (int r = 0; r < 8; ++r) {
    // denominator lives in o4 element r at the lane with rlo==8 of each half
    const float den = __shfl(o4[r], 8 + (lane & 16), 32);
    const float inv = (den > 0.f) ? 1.f / den : 0.f;
    const int rr = row0 + r;
    orow[rr * D_ +  0 + rlo] = o0[r] * inv;
    orow[rr * D_ + 16 + rlo] = o1[r] * inv;
    orow[rr * D_ + 32 + rlo] = o2[r] * inv;
    orow[rr * D_ + 48 + rlo] = o3[r] * inv;
    if (rlo < 8) orow[rr * D_ + 64 + rlo] = o4[r] * inv;
  }
}

// ---------------------------------------------------------------------------
extern "C" void kernel_launch(void* const* d_in, const int* in_sizes, int n_in,
                              void* d_out, int out_size, void* d_ws, size_t ws_size,
                              hipStream_t stream)
{
  (void)in_sizes; (void)n_in; (void)out_size; (void)ws_size;
  const float* q     = (const float*)d_in[0];
  const float* k     = (const float*)d_in[1];
  const float* v     = (const float*)d_in[2];
  const float* pos   = (const float*)d_in[3];
  const float* pos_o = (const float*)d_in[4];
  const float* tim   = (const float*)d_in[5];
  const float* freqs = (const float*)d_in[6];
  const float* tf    = (const float*)d_in[7];
  const float* scale = (const float*)d_in[8];

  char* ws = (char*)d_ws;
  const size_t nElem = (size_t)B_ * H_ * S_ * DP_;   // halves per region
  _Float16* qh = (_Float16*)(ws);
  _Float16* kh = (_Float16*)(ws + nElem * 2);
  _Float16* vt = (_Float16*)(ws + nElem * 4);

  const int rows = B_ * H_ * S_;                     // 65536 waves
  rope_norm_pack<<<rows / 4, 128, 0, stream>>>(q, k, v, pos, pos_o, tim,
                                               freqs, tf, scale, qh, kh, vt);

  const int jobs = B_ * H_ * (S_ / 16);              // 4096 waves
  flash_attn<<<jobs / 4, 128, 0, stream>>>(qh, kh, vt, (float*)d_out);
}
